// ScaledDotProductAttention_83013127897629
// MI455X (gfx1250) — matrix operand order repaired and checked by reading, not compiled, not run on hardware
//
#include <hip/hip_runtime.h>
#include <math.h>

// ---------------------------------------------------------------------------
// Shapes: B=8, S=2048, D=1024.  M = B*S = 16384 flattened rows.
// All inputs f32. Matmul operands bf16 for v_wmma_f32_16x16x32_bf16 (f32
// accumulate). Softmax / masks / residual in f32.
// gfx1250: WMMA bf16, double-buffered GLOBAL_LOAD_ASYNC_TO_LDS_B128 staging,
// s_wait_asynccnt, global_prefetch, register-pipelined fragment streams.
// ---------------------------------------------------------------------------

#define MROWS   16384
#define DDIM    1024
#define SEQ     2048
#define NEGC    (-4294967295.0f)   // -(2^32)+1
#define SCALE   0.03125f           // 1/sqrt(1024)
#define LSTRIDE 40                 // LDS row stride (elems): 80B rows, 16B aligned

typedef __attribute__((ext_vector_type(16))) __bf16 v16bf;
typedef __attribute__((ext_vector_type(8)))  float  v8f;

union Frag16 {
  v16bf v;
  unsigned int u[8];
};

__device__ __forceinline__ unsigned short f2bf(float f) {
  union { float f; unsigned int u; } x;
  x.f = f;
  unsigned int r = x.u + 0x7FFFu + ((x.u >> 16) & 1u);  // round-to-nearest-even
  return (unsigned short)(r >> 16);
}

// LDS byte offset of a generic pointer to __shared__ (flat LDS addr low 32 bits)
__device__ __forceinline__ unsigned lds_off32(const void* p) {
  return (unsigned)(unsigned long long)p;
}

// Async memory -> LDS copy, 16 bytes per issue.  Tracked by ASYNCcnt.
__device__ __forceinline__ void async_load_b128(unsigned lds_byte_off,
                                                const void* gaddr) {
  asm volatile("global_load_async_to_lds_b128 %0, %1, off"
               :: "v"(lds_byte_off), "v"((unsigned long long)gaddr)
               : "memory");
}

__device__ __forceinline__ void wait_asynccnt0() {
  asm volatile("s_wait_asynccnt 0x0" ::: "memory");
}

// 16 contiguous bf16 -> one fragment (2 x b128); works for LDS or global ptrs.
__device__ __forceinline__ void load_frag(Frag16& f,
                                          const unsigned short* base) {
  const uint4 u0 = *(const uint4*)base;
  const uint4 u1 = *(const uint4*)(base + 8);
  f.u[0] = u0.x; f.u[1] = u0.y; f.u[2] = u0.z; f.u[3] = u0.w;
  f.u[4] = u1.x; f.u[5] = u1.y; f.u[6] = u1.z; f.u[7] = u1.w;
}

// ---------------------------------------------------------------------------
// Kernel 1: masks[b,s] = sin(|sum_d x[b,s,d]|).  One wave32 per row.
// ---------------------------------------------------------------------------
__global__ void masks_kernel(const float* __restrict__ x,
                             float* __restrict__ masks) {
  const int wave = threadIdx.x >> 5;
  const int lane = threadIdx.x & 31;
  const int row  = blockIdx.x * 8 + wave;
  const float* p = x + (size_t)row * DDIM;
  float s = 0.f;
  #pragma unroll 8
  for (int i = 0; i < 32; ++i) s += p[lane + 32 * i];
  #pragma unroll
  for (int off = 16; off > 0; off >>= 1) s += __shfl_xor(s, off, 32);
  if (lane == 0) masks[row] = sinf(fabsf(s));
}

// ---------------------------------------------------------------------------
// Kernel 2: X (f32) -> Xbf (bf16), flat, vectorized.
// ---------------------------------------------------------------------------
__global__ void xconv_kernel(const float* __restrict__ in,
                             unsigned short* __restrict__ out) {
  const int i = blockIdx.x * blockDim.x + threadIdx.x;   // over float4 chunks
  const float4 f = ((const float4*)in)[i];
  ushort4 o;
  o.x = f2bf(f.x); o.y = f2bf(f.y); o.z = f2bf(f.z); o.w = f2bf(f.w);
  ((ushort4*)out)[i] = o;
}

// ---------------------------------------------------------------------------
// Kernel 3: W (1024x1024 f32, [k][n]) -> Wt (bf16, [n][k]) via LDS transpose.
// Block (32,8), grid (32,32).
// ---------------------------------------------------------------------------
__global__ void wtconv_kernel(const float* __restrict__ W,
                              unsigned short* __restrict__ Wt) {
  __shared__ float tile[32][33];
  const int tx = threadIdx.x, ty = threadIdx.y;
  const int r0 = blockIdx.y * 32, c0 = blockIdx.x * 32;
  #pragma unroll
  for (int i = 0; i < 4; ++i)
    tile[ty + 8 * i][tx] = W[(size_t)(r0 + ty + 8 * i) * DDIM + c0 + tx];
  __syncthreads();
  #pragma unroll
  for (int i = 0; i < 4; ++i)
    Wt[(size_t)(c0 + ty + 8 * i) * DDIM + r0 + tx] = f2bf(tile[tx][ty + 8 * i]);
}

// ---------------------------------------------------------------------------
// Kernel 4: projection GEMM  O = relu(Xbf @ Wt^T + b).
// A: [16384 x 1024] bf16 row-major, B: Wt [1024 cols x 1024 k] bf16.
// Block 128 threads (4 waves), 64x64 tile, K-step 32, double-buffered async
// LDS staging.  B fragments preloaded, 4 WMMAs issued back-to-back.
// MODE 0: Q (bf16 + f32)  1: K (bf16)  2: V (bf16 transposed V^T[b][d][key]).
// ---------------------------------------------------------------------------
template <int MODE>
__global__ void proj_gemm_kernel(const unsigned short* __restrict__ Abf,
                                 const unsigned short* __restrict__ Bt,
                                 const float* __restrict__ bias,
                                 unsigned short* __restrict__ Obf,
                                 float* __restrict__ Of32) {
  __shared__ __align__(16) unsigned short As[2][64 * LSTRIDE];
  __shared__ __align__(16) unsigned short Bs[2][64 * LSTRIDE];
  const int tid  = threadIdx.x;
  const int wave = tid >> 5;
  const int lane = tid & 31;
  const int lo16 = lane & 15;
  const int hi   = lane >> 4;
  const int row0 = blockIdx.y * 64;
  const int col0 = blockIdx.x * 64;

  auto stage = [&](int buf, int k0) {
    const unsigned aB = lds_off32(&As[buf][0]);
    const unsigned bB = lds_off32(&Bs[buf][0]);
    #pragma unroll
    for (int i = 0; i < 2; ++i) {
      const int c = i * 128 + tid;
      const int r = c >> 2, q = c & 3;
      async_load_b128(aB + r * (LSTRIDE * 2) + q * 16,
                      Abf + (size_t)(row0 + r) * DDIM + k0 + q * 8);
      async_load_b128(bB + r * (LSTRIDE * 2) + q * 16,
                      Bt + (size_t)(col0 + r) * DDIM + k0 + q * 8);
    }
  };

  v8f acc[4] = {};
  stage(0, 0);
  int cur = 0;
  for (int k0 = 0; k0 < DDIM; k0 += 32) {
    wait_asynccnt0();
    __syncthreads();
    if (k0 + 32 < DDIM) stage(cur ^ 1, k0 + 32);

    const unsigned short* Ab = &As[cur][0];
    const unsigned short* Bb = &Bs[cur][0];
    Frag16 a, b[4];
    load_frag(a, Ab + (wave * 16 + lo16) * LSTRIDE + hi * 16);
    #pragma unroll
    for (int t = 0; t < 4; ++t)
      load_frag(b[t], Bb + (t * 16 + lo16) * LSTRIDE + hi * 16);
    #pragma unroll
    for (int t = 0; t < 4; ++t)
      acc[t] = __builtin_amdgcn_wmma_f32_16x16x32_bf16(
          false, a.v, false, b[t].v, (short)0, acc[t], false, false);
    __syncthreads();
    cur ^= 1;
  }

  // Epilogue: bias + ReLU.  C/D layout: VGPR r -> row r (lanes<16) / r+8.
  const int rowM0 = row0 + wave * 16 + hi * 8;
  #pragma unroll
  for (int t = 0; t < 4; ++t) {
    const int colN = col0 + t * 16 + lo16;
    const float bv = bias[colN];
    #pragma unroll
    for (int r = 0; r < 8; ++r) {
      const int rowM = rowM0 + r;
      float v = acc[t][r] + bv;
      v = v > 0.f ? v : 0.f;
      if (MODE == 2) {
        const int bb = rowM >> 11, key = rowM & (SEQ - 1);
        Obf[(size_t)bb * DDIM * SEQ + (size_t)colN * SEQ + key] = f2bf(v);
      } else {
        const size_t o = (size_t)rowM * DDIM + colN;
        Obf[o] = f2bf(v);
        if (MODE == 0) Of32[o] = v;
      }
    }
  }
}

// ---------------------------------------------------------------------------
// Kernel 5: scores = Q K^T * SCALE, key-mask -> NEG, softmax over keys,
// attn *= qmask, P (bf16) -> scratch.  One wave per 16-query tile.
// Dynamic LDS: sQ (32KB, async-staged) + sS (16x2048 f32 = 128KB) = 160KB.
// Two key tiles per iteration (shared A, two acc chains) with register
// double-buffered fragment streams so K loads stay in flight across WMMAs.
// ---------------------------------------------------------------------------
__global__ void scores_softmax_kernel(const unsigned short* __restrict__ Qbf,
                                      const unsigned short* __restrict__ Kbf,
                                      const float* __restrict__ masks,
                                      unsigned short* __restrict__ P) {
  extern __shared__ __align__(16) char smem[];
  unsigned short* sQ = (unsigned short*)smem;                 // 16*1024 bf16
  float*          sS = (float*)(smem + 16 * DDIM * 2);        // 16*2048 f32

  const int lane = threadIdx.x & 31;
  const int lo16 = lane & 15;
  const int hi   = lane >> 4;
  const int q0   = blockIdx.x * 16;                 // flattened query row
  const int b    = q0 >> 11;
  const size_t keyRow0 = (size_t)b * SEQ;

  // Async-stage the Q tile (bf16 copy): 2048 x 16B chunks
  {
    const unsigned sqBase = lds_off32(sQ);
    const unsigned short* src = Qbf + (size_t)q0 * DDIM;
    #pragma unroll 4
    for (int i = 0; i < 64; ++i) {
      const int chunk = i * 32 + lane;
      async_load_b128(sqBase + chunk * 16, src + chunk * 8);
    }
    wait_asynccnt0();
  }
  __syncthreads();

  // Phase 1: 32 keys (2 WMMA tiles) per iteration
  const unsigned short* aRow = sQ + lo16 * DDIM + hi * 16;
  for (int kt2 = 0; kt2 < SEQ / 32; ++kt2) {
    v8f s0 = {}, s1 = {};
    const unsigned short* Kp0 =
        Kbf + (keyRow0 + (size_t)kt2 * 32 + lo16) * DDIM + hi * 16;
    const unsigned short* Kp1 = Kp0 + (size_t)16 * DDIM;
    if (kt2 + 1 < SEQ / 32) {
      __builtin_prefetch(Kp0 + (size_t)32 * DDIM, 0, 0);
      __builtin_prefetch(Kp1 + (size_t)32 * DDIM, 0, 0);
    }

    Frag16 a[2], f0[2], f1[2];
    load_frag(a[0], aRow);
    load_frag(f0[0], Kp0);
    load_frag(f1[0], Kp1);
    int cur = 0;
    #pragma unroll
    for (int k0 = 0; k0 < DDIM; k0 += 32) {
      const int nxt = cur ^ 1;
      if (k0 + 32 < DDIM) {
        load_frag(a[nxt], aRow + k0 + 32);
        load_frag(f0[nxt], Kp0 + k0 + 32);
        load_frag(f1[nxt], Kp1 + k0 + 32);
      }
      s0 = __builtin_amdgcn_wmma_f32_16x16x32_bf16(
          false, a[cur].v, false, f0[cur].v, (short)0, s0, false, false);
      s1 = __builtin_amdgcn_wmma_f32_16x16x32_bf16(
          false, a[cur].v, false, f1[cur].v, (short)0, s1, false, false);
      cur = nxt;
    }

    const int key0 = kt2 * 32 + lo16;
    const int key1 = key0 + 16;
    const float km0 = masks[keyRow0 + key0];
    const float km1 = masks[keyRow0 + key1];
    #pragma unroll
    for (int r = 0; r < 8; ++r) {
      const int rr = (r + hi * 8) * SEQ;
      sS[rr + key0] = (km0 == 0.0f) ? NEGC : s0[r] * SCALE;
      sS[rr + key1] = (km1 == 0.0f) ? NEGC : s1[r] * SCALE;
    }
  }
  __syncthreads();

  // Phase 2: softmax (2 lanes per row; each scans half the keys)
  {
    const int row = lo16;
    float* rp = sS + row * SEQ + hi * 1024;
    float mx = -3.4e38f;
    for (int i = 0; i < 1024; ++i) mx = fmaxf(mx, rp[i]);
    mx = fmaxf(mx, __shfl_xor(mx, 16, 32));
    float sum = 0.f;
    for (int i = 0; i < 1024; ++i) {
      float p = __expf(rp[i] - mx);
      sum += p;
      rp[i] = p;
    }
    sum += __shfl_xor(sum, 16, 32);
    const float qm  = masks[q0 + row];
    const float inv = qm / sum;                     // fold post-softmax q-mask
    unsigned int* op = (unsigned int*)(P + (size_t)(q0 + row) * SEQ + hi * 1024);
    for (int i = 0; i < 512; ++i) {
      const unsigned short p0 = f2bf(rp[2 * i] * inv);
      const unsigned short p1 = f2bf(rp[2 * i + 1] * inv);
      op[i] = (unsigned)p0 | ((unsigned)p1 << 16);
    }
  }
}

// ---------------------------------------------------------------------------
// Kernel 6: out = P @ V + Q.  M=16384, N=1024, K=2048.
// Both tiles async-staged (V pre-transposed: V^T[b][d][key]), double-
// buffered; residual Q folded into the accumulator init; 4 WMMAs issued
// back-to-back per K-step.
// ---------------------------------------------------------------------------
__global__ void pv_gemm_kernel(const unsigned short* __restrict__ P,
                               const unsigned short* __restrict__ VbfT,
                               const float* __restrict__ Qf32,
                               float* __restrict__ out) {
  __shared__ __align__(16) unsigned short As[2][64 * LSTRIDE];
  __shared__ __align__(16) unsigned short Bs[2][64 * LSTRIDE];
  const int tid  = threadIdx.x;
  const int wave = tid >> 5;
  const int lane = tid & 31;
  const int lo16 = lane & 15;
  const int hi   = lane >> 4;
  const int row0 = blockIdx.y * 64;                 // query rows
  const int col0 = blockIdx.x * 64;                 // feature cols
  const int b    = row0 >> 11;
  const unsigned short* VT = VbfT + (size_t)b * DDIM * SEQ;

  auto stage = [&](int buf, int k0) {
    const unsigned aB = lds_off32(&As[buf][0]);
    const unsigned bB = lds_off32(&Bs[buf][0]);
    #pragma unroll
    for (int i = 0; i < 2; ++i) {
      const int c = i * 128 + tid;
      const int r = c >> 2, q = c & 3;
      async_load_b128(aB + r * (LSTRIDE * 2) + q * 16,
                      P + (size_t)(row0 + r) * SEQ + k0 + q * 8);
      async_load_b128(bB + r * (LSTRIDE * 2) + q * 16,
                      VT + (size_t)(col0 + r) * SEQ + k0 + q * 8);
    }
  };

  // Residual: accumulator starts at Q (f32)
  v8f acc[4];
  #pragma unroll
  for (int t = 0; t < 4; ++t) {
    #pragma unroll
    for (int r = 0; r < 8; ++r) {
      acc[t][r] = Qf32[(size_t)(row0 + wave * 16 + r + hi * 8) * DDIM
                       + col0 + t * 16 + lo16];
    }
  }

  stage(0, 0);
  int cur = 0;
  for (int k0 = 0; k0 < SEQ; k0 += 32) {
    wait_asynccnt0();
    __syncthreads();
    if (k0 + 32 < SEQ) stage(cur ^ 1, k0 + 32);

    const unsigned short* Ab = &As[cur][0];
    const unsigned short* Bb = &Bs[cur][0];
    Frag16 a, bfr[4];
    load_frag(a, Ab + (wave * 16 + lo16) * LSTRIDE + hi * 16);
    #pragma unroll
    for (int t = 0; t < 4; ++t)
      load_frag(bfr[t], Bb + (t * 16 + lo16) * LSTRIDE + hi * 16);
    #pragma unroll
    for (int t = 0; t < 4; ++t)
      acc[t] = __builtin_amdgcn_wmma_f32_16x16x32_bf16(
          false, a.v, false, bfr[t].v, (short)0, acc[t], false, false);
    __syncthreads();
    cur ^= 1;
  }

  const size_t rbase = (size_t)(row0 + wave * 16 + hi * 8) * DDIM;
  #pragma unroll
  for (int t = 0; t < 4; ++t) {
    const int colN = col0 + t * 16 + lo16;
    size_t o = rbase + colN;
    #pragma unroll
    for (int r = 0; r < 8; ++r, o += DDIM) {
      out[o] = acc[t][r];
    }
  }
}

// ---------------------------------------------------------------------------
// Host-side launcher.  d_ws layout (bytes):
//   [0, 64K)   masks (16384 f32)
//   + 64M      Qf32
//   + 32M      Qbf        + 32M   Kbf        + 32M   VbfT (transposed)
//   + 64M      P (16384x2048 bf16)
//   + 32M      Xbf
//   + 3 x 2M   Wqt / Wkt / Wvt (bf16, pre-transposed)
// ---------------------------------------------------------------------------
extern "C" void kernel_launch(void* const* d_in, const int* in_sizes, int n_in,
                              void* d_out, int out_size, void* d_ws, size_t ws_size,
                              hipStream_t stream) {
  const float* x  = (const float*)d_in[0];
  const float* Wq = (const float*)d_in[1];
  const float* bq = (const float*)d_in[2];
  const float* Wk = (const float*)d_in[3];
  const float* bk = (const float*)d_in[4];
  const float* Wv = (const float*)d_in[5];
  const float* bv = (const float*)d_in[6];
  float* out = (float*)d_out;

  char* ws = (char*)d_ws;
  const size_t nQK = (size_t)MROWS * DDIM;            // 16M elements
  const size_t nW  = (size_t)DDIM * DDIM;             // 1M elements
  float* masks = (float*)ws;
  float* Qf32  = (float*)(ws + 65536);
  unsigned short* Qbf  = (unsigned short*)(ws + 65536 + nQK * sizeof(float));
  unsigned short* Kbf  = Qbf + nQK;
  unsigned short* VbfT = Kbf + nQK;
  unsigned short* P    = VbfT + nQK;                  // 16384*2048 bf16
  unsigned short* Xbf  = P + (size_t)MROWS * SEQ;
  unsigned short* Wqt  = Xbf + nQK;
  unsigned short* Wkt  = Wqt + nW;
  unsigned short* Wvt  = Wkt + nW;

  // 1) masks + operand conversion
  masks_kernel<<<MROWS / 8, 256, 0, stream>>>(x, masks);
  xconv_kernel<<<(int)(nQK / 4 / 256), 256, 0, stream>>>(x, Xbf);
  dim3 tgrid(32, 32), tblk(32, 8);
  wtconv_kernel<<<tgrid, tblk, 0, stream>>>(Wq, Wqt);
  wtconv_kernel<<<tgrid, tblk, 0, stream>>>(Wk, Wkt);
  wtconv_kernel<<<tgrid, tblk, 0, stream>>>(Wv, Wvt);

  // 2) QKV projections (Q keeps f32 for the residual; V stored transposed)
  dim3 ggrid(DDIM / 64, MROWS / 64);
  proj_gemm_kernel<0><<<ggrid, 128, 0, stream>>>(Xbf, Wqt, bq, Qbf, Qf32);
  proj_gemm_kernel<1><<<ggrid, 128, 0, stream>>>(Xbf, Wkt, bk, Kbf, nullptr);
  proj_gemm_kernel<2><<<ggrid, 128, 0, stream>>>(Xbf, Wvt, bv, VbfT, nullptr);

  // 3) scores + masked softmax -> P (bf16)
  const size_t smem = (size_t)16 * DDIM * 2 + (size_t)16 * SEQ * 4;  // 160KB
  scores_softmax_kernel<<<MROWS / 16, 32, smem, stream>>>(Qbf, Kbf, masks, P);

  // 4) out = P @ V + Q
  pv_gemm_kernel<<<ggrid, 128, 0, stream>>>(P, VbfT, Qf32, out);
}
